// EGNN_10411000725826
// MI455X (gfx1250) — matrix operand order repaired and checked by reading, not compiled, not run on hardware
//
#include <hip/hip_runtime.h>
#include <hip/hip_bf16.h>

#define NN 40000
#define EE 640000
#define HH 128

// LDS row strides (bf16 elems), padded for conflict-free ds_load_b128:
// EIN_S = 296 -> 148 dwords ( ≡20 mod 64, 4*5r pattern, full bank cover)
// EFL_S = 136 ->  68 dwords ( ≡ 4 mod 64, perfect cover)
#define EIN_S 296
#define EFL_S 136

typedef __attribute__((ext_vector_type(16))) __bf16 v16bf;
typedef __attribute__((ext_vector_type(8)))  float  v8f;
typedef __attribute__((ext_vector_type(8)))  unsigned int v8u;

// ---------- helpers ----------

__device__ __forceinline__ unsigned short f2bf(float f) {
    unsigned int u = __builtin_bit_cast(unsigned int, f);
    unsigned int r = u + 0x7fffu + ((u >> 16) & 1u);
    return (unsigned short)(r >> 16);
}

__device__ __forceinline__ float silu_f(float x) {
    return x / (1.f + __expf(-x));
}

// A-fragment (16x32 bf16, ISA 7.12.2): lane holds row M=lane&15,
// elems 0..7 = K kbase..+7, elems 8..15 = K kbase+16..+23, kbase=(lane>>4)*8.
__device__ __forceinline__ v16bf frag_gap(const unsigned short* p) {
    uint4 q0 = *(const uint4*)(p);
    uint4 q1 = *(const uint4*)(p + 16);
    v8u t;
    t[0] = q0.x; t[1] = q0.y; t[2] = q0.z; t[3] = q0.w;
    t[4] = q1.x; t[5] = q1.y; t[6] = q1.z; t[7] = q1.w;
    return __builtin_bit_cast(v16bf, t);
}

// B-fragment from packed weights: 32 contiguous bytes per lane.
__device__ __forceinline__ v16bf frag_cont(const unsigned short* p) {
    uint4 q0 = *(const uint4*)(p);
    uint4 q1 = *(const uint4*)(p + 8);
    v8u t;
    t[0] = q0.x; t[1] = q0.y; t[2] = q0.z; t[3] = q0.w;
    t[4] = q1.x; t[5] = q1.y; t[6] = q1.z; t[7] = q1.w;
    return __builtin_bit_cast(v16bf, t);
}

__device__ __forceinline__ v8f wmma_bf16(v16bf a, v16bf b, v8f c) {
    return __builtin_amdgcn_wmma_f32_16x16x32_bf16(false, a, false, b, (short)0, c,
                                                   false, false);
}

__device__ __forceinline__ const unsigned short*
bptr(const unsigned short* Wp, int kt, int w, int ntiles, int lane) {
    return Wp + ((size_t)(kt * ntiles + w) * 32 + lane) * 16;
}

// ---------- weight packing ----------
// dst layout: [ktile][ntile][lane][16] bf16 elements.
// n = ntile*16 + (lane&15) ; k = ktile*32 + (lane>>4)*8 + (j&7) + (j>>3)*16
__global__ void pack_w(const float* __restrict__ W, unsigned short* __restrict__ dst,
                       int Korig, int Norig, int ntiles, int total) {
    for (int idx = blockIdx.x * blockDim.x + threadIdx.x; idx < total;
         idx += gridDim.x * blockDim.x) {
        int j    = idx & 15;
        int lane = (idx >> 4) & 31;
        int blk  = idx >> 9;
        int nt   = blk % ntiles;
        int kt   = blk / ntiles;
        int n = nt * 16 + (lane & 15);
        int k = kt * 32 + ((lane >> 4) << 3) + (j & 7) + ((j >> 3) << 4);
        float v = (k < Korig && n < Norig) ? W[(size_t)k * Norig + n] : 0.f;
        dst[idx] = f2bf(v);
    }
}

// ---------- embedding: h = h0 @ emb_w + emb_b ----------
__global__ void embed_k(const float* __restrict__ h0, const float* __restrict__ Wemb,
                        const float* __restrict__ bemb, float* __restrict__ h,
                        unsigned short* __restrict__ hbf) {
    int idx = blockIdx.x * blockDim.x + threadIdx.x;
    if (idx >= NN * HH) return;
    int n = idx >> 7, c = idx & 127;
    float a = bemb[c];
#pragma unroll
    for (int k = 0; k < 15; ++k) a += h0[n * 15 + k] * Wemb[k * HH + c];
    h[idx]   = a;
    hbf[idx] = f2bf(a);
}

// ---------- radial ----------
__global__ void radial_k(const float* __restrict__ x, const int* __restrict__ edges,
                         float* __restrict__ radial) {
    int e = blockIdx.x * blockDim.x + threadIdx.x;
    if (e >= EE) return;
    int r = edges[e], c = edges[EE + e];
    float dx = x[r * 3 + 0] - x[c * 3 + 0];
    float dy = x[r * 3 + 1] - x[c * 3 + 1];
    float dz = x[r * 3 + 2] - x[c * 3 + 2];
    radial[e] = dx * dx + dy * dy + dz * dz;
}

// ---------- fused edge model: 64 edges per block, 8 waves, 4 M-subtiles ----------
__global__ void __launch_bounds__(256)
edge_fused(const unsigned short* __restrict__ hbf, const float* __restrict__ radial,
           const float* __restrict__ eattr, const int* __restrict__ edges,
           const unsigned short* __restrict__ W1p, const float* __restrict__ b1,
           const unsigned short* __restrict__ W2p, const float* __restrict__ b2,
           float* __restrict__ agg) {
    __shared__ __align__(16) unsigned short ein[64][EIN_S];
    __shared__ __align__(16) unsigned short efl[64][EFL_S];
    __shared__ int rows_s[64];

    const int tid = threadIdx.x;
    const int eb  = blockIdx.x * 64;

    // meta: radial, edge_attr, zero pad (K 256..287)
    if (tid < 64) {
        int e = eb + tid;
        rows_s[tid]   = edges[e];
        ein[tid][256] = f2bf(radial[e]);
        const float* ea = eattr + (size_t)e * 4;
        ein[tid][257] = f2bf(ea[0]);
        ein[tid][258] = f2bf(ea[1]);
        ein[tid][259] = f2bf(ea[2]);
        ein[tid][260] = f2bf(ea[3]);
        for (int k = 261; k < 288; ++k) ein[tid][k] = 0;
    }
    // gather h[row] (K 0..127) and h[col] (K 128..255): 64 edges x 16 chunks
#pragma unroll
    for (int it = 0; it < 4; ++it) {
        int t2 = tid + it * 256;
        int j = t2 >> 4, p = t2 & 15;
        int e = eb + j;
        int src = (p < 8) ? edges[e] : edges[EE + e];
        int c0  = (p & 7) * 16;
        const uint4* s = (const uint4*)(hbf + (size_t)src * HH + c0);
        uint4 a0 = s[0], a1 = s[1];
        uint4* d = (uint4*)&ein[j][((p < 8) ? 0 : 128) + c0];
        d[0] = a0; d[1] = a1;
    }
    __syncthreads();

    const int lane = tid & 31;
    const int w    = tid >> 5;
    const int arow = lane & 15;
    const int kb   = (lane >> 4) * 8;
    const int mofs = (lane >> 4) << 3;
    const int n    = w * 16 + (lane & 15);

    v8f acc[4];
#pragma unroll
    for (int s = 0; s < 4; ++s)
#pragma unroll
        for (int m = 0; m < 8; ++m) acc[s][m] = 0.f;
#pragma unroll
    for (int kt = 0; kt < 9; ++kt) {
        v16bf b = frag_cont(bptr(W1p, kt, w, 8, lane));
#pragma unroll
        for (int s = 0; s < 4; ++s) {
            v16bf a = frag_gap(&ein[s * 16 + arow][kt * 32 + kb]);
            acc[s] = wmma_bf16(a, b, acc[s]);
        }
    }
    float bias1 = b1[n];
#pragma unroll
    for (int s = 0; s < 4; ++s)
#pragma unroll
        for (int m = 0; m < 8; ++m)
            efl[s * 16 + m + mofs][n] = f2bf(silu_f(acc[s][m] + bias1));
    __syncthreads();

    v8f acc2[4];
#pragma unroll
    for (int s = 0; s < 4; ++s)
#pragma unroll
        for (int m = 0; m < 8; ++m) acc2[s][m] = 0.f;
#pragma unroll
    for (int kt = 0; kt < 4; ++kt) {
        v16bf b = frag_cont(bptr(W2p, kt, w, 8, lane));
#pragma unroll
        for (int s = 0; s < 4; ++s) {
            v16bf a = frag_gap(&efl[s * 16 + arow][kt * 32 + kb]);
            acc2[s] = wmma_bf16(a, b, acc2[s]);
        }
    }
    float bias2 = b2[n];
#pragma unroll
    for (int s = 0; s < 4; ++s)
#pragma unroll
        for (int m = 0; m < 8; ++m) {
            float v  = silu_f(acc2[s][m] + bias2);
            int node = rows_s[s * 16 + m + mofs];
            atomicAdd(&agg[(size_t)node * HH + n], v);
        }
}

// ---------- fused node model: 64 nodes per block, residual update ----------
__global__ void __launch_bounds__(256)
node_fused(unsigned short* __restrict__ hbf, const float* __restrict__ agg,
           const float* __restrict__ h0,
           const unsigned short* __restrict__ W1p, const float* __restrict__ b1,
           const unsigned short* __restrict__ W2p, const float* __restrict__ b2,
           float* __restrict__ h) {
    __shared__ __align__(16) unsigned short ein[64][EIN_S];
    __shared__ __align__(16) unsigned short efl[64][EFL_S];

    const int tid = threadIdx.x;
    const int nb  = blockIdx.x * 64;

    // fill [h | agg]: 64 nodes x 16 chunks of 16
#pragma unroll
    for (int it = 0; it < 4; ++it) {
        int t2 = tid + it * 256;
        int j = t2 >> 4, p = t2 & 15;
        int node = nb + j;
        if (p < 8) {
            int c0 = p * 16;
            const uint4* s = (const uint4*)(hbf + (size_t)node * HH + c0);
            uint4 a0 = s[0], a1 = s[1];
            uint4* d = (uint4*)&ein[j][c0];
            d[0] = a0; d[1] = a1;
        } else {
            int c0 = (p - 8) * 16;
            for (int q = 0; q < 16; ++q)
                ein[j][128 + c0 + q] = f2bf(agg[(size_t)node * HH + c0 + q]);
        }
    }
    // fill [h0 | pad]: 64 nodes x 2 chunks
    if (tid < 128) {
        int j = tid >> 1, p = tid & 1;
        int node = nb + j;
        int base = 256 + p * 16;
        for (int q = 0; q < 16; ++q) {
            int k = base + q;
            float v = (k < 271) ? h0[(size_t)node * 15 + (k - 256)] : 0.f;
            ein[j][k] = f2bf(v);
        }
    }
    __syncthreads();

    const int lane = tid & 31;
    const int w    = tid >> 5;
    const int arow = lane & 15;
    const int kb   = (lane >> 4) * 8;
    const int mofs = (lane >> 4) << 3;
    const int n    = w * 16 + (lane & 15);

    v8f acc[4];
#pragma unroll
    for (int s = 0; s < 4; ++s)
#pragma unroll
        for (int m = 0; m < 8; ++m) acc[s][m] = 0.f;
#pragma unroll
    for (int kt = 0; kt < 9; ++kt) {
        v16bf b = frag_cont(bptr(W1p, kt, w, 8, lane));
#pragma unroll
        for (int s = 0; s < 4; ++s) {
            v16bf a = frag_gap(&ein[s * 16 + arow][kt * 32 + kb]);
            acc[s] = wmma_bf16(a, b, acc[s]);
        }
    }
    float bias1 = b1[n];
#pragma unroll
    for (int s = 0; s < 4; ++s)
#pragma unroll
        for (int m = 0; m < 8; ++m)
            efl[s * 16 + m + mofs][n] = f2bf(silu_f(acc[s][m] + bias1));
    __syncthreads();

    v8f acc2[4];
#pragma unroll
    for (int s = 0; s < 4; ++s)
#pragma unroll
        for (int m = 0; m < 8; ++m) acc2[s][m] = 0.f;
#pragma unroll
    for (int kt = 0; kt < 4; ++kt) {
        v16bf b = frag_cont(bptr(W2p, kt, w, 8, lane));
#pragma unroll
        for (int s = 0; s < 4; ++s) {
            v16bf a = frag_gap(&efl[s * 16 + arow][kt * 32 + kb]);
            acc2[s] = wmma_bf16(a, b, acc2[s]);
        }
    }
    float bias2 = b2[n];
#pragma unroll
    for (int s = 0; s < 4; ++s)
#pragma unroll
        for (int m = 0; m < 8; ++m) {
            int grow = nb + s * 16 + m + mofs;
            float v = h[(size_t)grow * HH + n] + acc2[s][m] + bias2;  // residual
            h[(size_t)grow * HH + n]   = v;
            hbf[(size_t)grow * HH + n] = f2bf(v);
        }
}

// ---------- fused decoder pair: out = silu(X@W1+b1)@W2 + b2, 64 rows/block ----------
__global__ void __launch_bounds__(256)
dec_fused(const unsigned short* __restrict__ X,
          const unsigned short* __restrict__ W1p, const float* __restrict__ b1,
          const unsigned short* __restrict__ W2p, const float* __restrict__ b2,
          int n2tiles, int nwrite, float* __restrict__ outF,
          unsigned short* __restrict__ outB) {
    __shared__ __align__(16) unsigned short efl[64][EFL_S];

    const int tid  = threadIdx.x;
    const int nb   = blockIdx.x * 64;
    const int lane = tid & 31;
    const int w    = tid >> 5;
    const int arow = lane & 15;
    const int kb   = (lane >> 4) * 8;
    const int mofs = (lane >> 4) << 3;
    const int n    = w * 16 + (lane & 15);

    v8f acc[4];
#pragma unroll
    for (int s = 0; s < 4; ++s)
#pragma unroll
        for (int m = 0; m < 8; ++m) acc[s][m] = 0.f;
#pragma unroll
    for (int kt = 0; kt < 4; ++kt) {
        v16bf b = frag_cont(bptr(W1p, kt, w, 8, lane));
#pragma unroll
        for (int s = 0; s < 4; ++s) {
            v16bf a = frag_gap(X + (size_t)(nb + s * 16 + arow) * HH + kt * 32 + kb);
            acc[s] = wmma_bf16(a, b, acc[s]);
        }
    }
    float bias1 = b1[n];
#pragma unroll
    for (int s = 0; s < 4; ++s)
#pragma unroll
        for (int m = 0; m < 8; ++m)
            efl[s * 16 + m + mofs][n] = f2bf(silu_f(acc[s][m] + bias1));
    __syncthreads();

    if (w < n2tiles) {
        v8f acc2[4];
#pragma unroll
        for (int s = 0; s < 4; ++s)
#pragma unroll
            for (int m = 0; m < 8; ++m) acc2[s][m] = 0.f;
#pragma unroll
        for (int kt = 0; kt < 4; ++kt) {
            v16bf b = frag_cont(bptr(W2p, kt, w, n2tiles, lane));
#pragma unroll
            for (int s = 0; s < 4; ++s) {
                v16bf a = frag_gap(&efl[s * 16 + arow][kt * 32 + kb]);
                acc2[s] = wmma_bf16(a, b, acc2[s]);
            }
        }
        float bias2 = (n < nwrite) ? b2[n] : 0.f;
#pragma unroll
        for (int s = 0; s < 4; ++s)
#pragma unroll
            for (int m = 0; m < 8; ++m) {
                int grow = nb + s * 16 + m + mofs;
                float v  = acc2[s][m] + bias2;
                if (outF && n < nwrite) outF[(size_t)grow * nwrite + n] = v;
                if (outB) outB[(size_t)grow * HH + n] = f2bf(v);
            }
    }
}

// ---------- host launch ----------
extern "C" void kernel_launch(void* const* d_in, const int* in_sizes, int n_in,
                              void* d_out, int out_size, void* d_ws, size_t ws_size,
                              hipStream_t stream) {
    (void)in_sizes; (void)n_in; (void)out_size; (void)ws_size;

    const float* h0    = (const float*)d_in[0];
    const float* x     = (const float*)d_in[1];
    const int*   edges = (const int*)d_in[2];
    const float* eattr = (const float*)d_in[3];
    const float* emb_w = (const float*)d_in[4];
    const float* emb_b = (const float*)d_in[5];
    const float* ew1   = (const float*)d_in[6];
    const float* eb1   = (const float*)d_in[7];
    const float* ew2   = (const float*)d_in[8];
    const float* eb2   = (const float*)d_in[9];
    const float* nw1   = (const float*)d_in[10];
    const float* nb1   = (const float*)d_in[11];
    const float* nw2   = (const float*)d_in[12];
    const float* nb2   = (const float*)d_in[13];
    const float* dw1   = (const float*)d_in[14];
    const float* db1   = (const float*)d_in[15];
    const float* dw2   = (const float*)d_in[16];
    const float* db2   = (const float*)d_in[17];
    const float* gw1   = (const float*)d_in[18];
    const float* gb1   = (const float*)d_in[19];
    const float* gw2   = (const float*)d_in[20];
    const float* gb2   = (const float*)d_in[21];

    // workspace carve
    size_t off = 0;
    char* base = (char*)d_ws;
    auto carve = [&](size_t bytes) -> char* {
        char* p = base + off;
        off += (bytes + 255) & ~(size_t)255;
        return p;
    };
    float*          h      = (float*)carve((size_t)NN * HH * 4);
    float*          agg    = (float*)carve((size_t)NN * HH * 4);
    float*          radial = (float*)carve((size_t)EE * 4);
    unsigned short* hbf    = (unsigned short*)carve((size_t)NN * HH * 2);
    unsigned short* bufA   = (unsigned short*)carve((size_t)NN * HH * 2);
    unsigned short* wts    = (unsigned short*)carve((size_t)479232 * 2);

    // packed weight offsets (bf16 elements)
    const int OFF_EW1 = 0;       // + i*36864
    const int OFF_EW2 = 147456;  // + i*16384
    const int OFF_NW1 = 212992;  // + i*36864
    const int OFF_NW2 = 360448;  // + i*16384
    const int OFF_DW1 = 425984;
    const int OFF_DW2 = 442368;
    const int OFF_GW1 = 458752;
    const int OFF_GW2 = 475136;

    auto packLaunch = [&](const float* W, unsigned short* dst, int Ko, int Kp,
                          int No, int Np) {
        int total = (Kp / 32) * (Np / 16) * 512;
        pack_w<<<(total + 255) / 256, 256, 0, stream>>>(W, dst, Ko, No, Np / 16, total);
    };
    for (int i = 0; i < 4; ++i) {
        packLaunch(ew1 + (size_t)i * 261 * HH, wts + OFF_EW1 + i * 36864, 261, 288, HH, HH);
        packLaunch(ew2 + (size_t)i * HH * HH,  wts + OFF_EW2 + i * 16384, HH, HH, HH, HH);
        packLaunch(nw1 + (size_t)i * 271 * HH, wts + OFF_NW1 + i * 36864, 271, 288, HH, HH);
        packLaunch(nw2 + (size_t)i * HH * HH,  wts + OFF_NW2 + i * 16384, HH, HH, HH, HH);
    }
    packLaunch(dw1, wts + OFF_DW1, HH, HH, HH, HH);
    packLaunch(dw2, wts + OFF_DW2, HH, HH, HH, HH);
    packLaunch(gw1, wts + OFF_GW1, HH, HH, HH, HH);
    packLaunch(gw2, wts + OFF_GW2, HH, HH, 21, 32);

    embed_k<<<(NN * HH + 255) / 256, 256, 0, stream>>>(h0, emb_w, emb_b, h, hbf);
    radial_k<<<(EE + 255) / 256, 256, 0, stream>>>(x, edges, radial);

    for (int i = 0; i < 4; ++i) {
        hipMemsetAsync(agg, 0, (size_t)NN * HH * 4, stream);
        edge_fused<<<EE / 64, 256, 0, stream>>>(
            hbf, radial, eattr, edges,
            wts + OFF_EW1 + i * 36864, eb1 + (size_t)i * HH,
            wts + OFF_EW2 + i * 16384, eb2 + (size_t)i * HH, agg);
        node_fused<<<NN / 64, 256, 0, stream>>>(
            hbf, agg, h0,
            wts + OFF_NW1 + i * 36864, nb1 + (size_t)i * HH,
            wts + OFF_NW2 + i * 16384, nb2 + (size_t)i * HH, h);
    }

    // node_dec:  t2 = silu(h@dw1+db1)@dw2 + db2   -> bufA (bf16)
    dec_fused<<<NN / 64, 256, 0, stream>>>(hbf, wts + OFF_DW1, db1,
                                           wts + OFF_DW2, db2, 8, 128, nullptr, bufA);
    // graph_dec: pred = silu(t2@gw1+gb1)@gw2 + gb2 -> d_out [N,21]
    dec_fused<<<NN / 64, 256, 0, stream>>>(bufA, wts + OFF_GW1, gb1,
                                           wts + OFF_GW2, gb2, 2, 21, (float*)d_out,
                                           nullptr);
}
